// HybridBinaryClassifier_65481071405658
// MI455X (gfx1250) — matrix operand order repaired and checked by reading, not compiled, not run on hardware
//
#include <hip/hip_runtime.h>
#include <hip/hip_bf16.h>
#include <math.h>

// ---------------------------------------------------------------------------
// HybridBinaryClassifier on gfx1250 (MI455X), wave32 + WMMA + async-to-LDS.
// Pipeline:
//   k1: conv1(5x5,s2,p1)+bias+relu+maxpool2x2s1  -> h1 [256,6,123,123] f16
//   k2: conv2(3x3,s2,p1)+bias+relu+maxpool2x2s1  -> h2 [256,55840] f16 (padded K)
//   k3: zero K-padding of h2
//   k4: fc1_w f32[120,55815] -> f16[128,55840] (zero padded)
//   k5: fc1 GEMM: async global->LDS double buffer + v_wmma_f32_16x16x32_f16
//   k6: fc2 + norms + fc3 + tanh head + sigmoid -> features/norms/probs
//   k7: cosine-sim graph, degree-weighted aggregation -> out [256,2]
// ---------------------------------------------------------------------------

typedef __attribute__((ext_vector_type(16))) _Float16 v16h;
typedef __attribute__((ext_vector_type(8)))  float    v8f;
typedef __attribute__((ext_vector_type(4)))  unsigned int u32x4;
typedef int b128i __attribute__((vector_size(16)));   // matches builtin param type

#define BATCH   256
#define KFEAT   55815     // 15*61*61
#define KPAD    55840     // next multiple of 32
#define NPAD    128       // fc1 out 120 padded to 128

union FragF16 {
    v16h  h;
    u32x4 q[2];
};

#if defined(__gfx1250__) && __has_builtin(__builtin_amdgcn_global_load_async_to_lds_b128)
#define HAS_ASYNC_LDS 1
#else
#define HAS_ASYNC_LDS 0
#endif

#if HAS_ASYNC_LDS
// 16-byte async copy global -> LDS, tracked by ASYNCcnt.
// Builtin signature (from compiler diagnostic): (v4i AS1*, v4i AS3*, imm, imm).
__device__ __forceinline__ void async_cp16(const _Float16* g, _Float16* l) {
    __builtin_amdgcn_global_load_async_to_lds_b128(
        (__attribute__((address_space(1))) b128i*)(unsigned long long)g,
        (__attribute__((address_space(3))) b128i*)(unsigned int)(unsigned long long)l,
        0, 0);
}
#endif

// ---------------- conv1 + relu + maxpool(2,2,s1) --------------------------
// out: h1[b][c][y][x], y,x in [0,123), f16.  23,238,144 threads.
__global__ void k_conv1_pool(const float* __restrict__ x,
                             const float* __restrict__ w,
                             const float* __restrict__ bias,
                             _Float16* __restrict__ h1) {
    int idx = blockIdx.x * blockDim.x + threadIdx.x;
    int xw = idx % 123; int t = idx / 123;
    int yh = t % 123;   t /= 123;
    int c  = t % 6;     int b = t / 6;

    float s00 = 0.f, s01 = 0.f, s10 = 0.f, s11 = 0.f;
    for (int ci = 0; ci < 3; ++ci) {
        const float* xp = x + ((size_t)(b * 3 + ci)) * 250 * 250;
        const float* wp = w + ((c * 3 + ci) * 25);
        #pragma unroll
        for (int ky = 0; ky < 5; ++ky) {
            #pragma unroll
            for (int kx = 0; kx < 5; ++kx) {
                float wv = wp[ky * 5 + kx];
                int iy0 = 2 * yh + ky - 1;
                int ix0 = 2 * xw + kx - 1;
                float v00 = (iy0 >= 0 && iy0 < 250 && ix0 >= 0 && ix0 < 250)
                              ? xp[iy0 * 250 + ix0] : 0.f;
                float v01 = (iy0 >= 0 && iy0 < 250 && ix0 + 2 < 250)
                              ? xp[iy0 * 250 + ix0 + 2] : 0.f;
                float v10 = (iy0 + 2 < 250 && ix0 >= 0 && ix0 < 250)
                              ? xp[(iy0 + 2) * 250 + ix0] : 0.f;
                float v11 = (iy0 + 2 < 250 && ix0 + 2 < 250)
                              ? xp[(iy0 + 2) * 250 + ix0 + 2] : 0.f;
                s00 += wv * v00; s01 += wv * v01;
                s10 += wv * v10; s11 += wv * v11;
            }
        }
    }
    float bb = bias[c];
    float r00 = fmaxf(s00 + bb, 0.f), r01 = fmaxf(s01 + bb, 0.f);
    float r10 = fmaxf(s10 + bb, 0.f), r11 = fmaxf(s11 + bb, 0.f);
    float m = fmaxf(fmaxf(r00, r01), fmaxf(r10, r11));
    h1[idx] = (_Float16)m;
}

// ---------------- conv2 + relu + maxpool(2,2,s1) --------------------------
// out: h2[b][c*3721 + y*61 + x] f16 (row stride KPAD). 14,288,640 threads.
__global__ void k_conv2_pool(const _Float16* __restrict__ h1,
                             const float* __restrict__ w,
                             const float* __restrict__ bias,
                             _Float16* __restrict__ h2) {
    int idx = blockIdx.x * blockDim.x + threadIdx.x;
    int xw = idx % 61; int t = idx / 61;
    int yh = t % 61;   t /= 61;
    int c  = t % 15;   int b = t / 15;

    float s00 = 0.f, s01 = 0.f, s10 = 0.f, s11 = 0.f;
    for (int ci = 0; ci < 6; ++ci) {
        const _Float16* hp = h1 + ((size_t)(b * 6 + ci)) * 123 * 123;
        const float* wp = w + ((c * 6 + ci) * 9);
        #pragma unroll
        for (int ky = 0; ky < 3; ++ky) {
            #pragma unroll
            for (int kx = 0; kx < 3; ++kx) {
                float wv = wp[ky * 3 + kx];
                int iy0 = 2 * yh + ky - 1;
                int ix0 = 2 * xw + kx - 1;
                float v00 = (iy0 >= 0 && iy0 < 123 && ix0 >= 0 && ix0 < 123)
                              ? (float)hp[iy0 * 123 + ix0] : 0.f;
                float v01 = (iy0 >= 0 && iy0 < 123 && ix0 + 2 < 123)
                              ? (float)hp[iy0 * 123 + ix0 + 2] : 0.f;
                float v10 = (iy0 + 2 < 123 && ix0 >= 0 && ix0 < 123)
                              ? (float)hp[(iy0 + 2) * 123 + ix0] : 0.f;
                float v11 = (iy0 + 2 < 123 && ix0 + 2 < 123)
                              ? (float)hp[(iy0 + 2) * 123 + ix0 + 2] : 0.f;
                s00 += wv * v00; s01 += wv * v01;
                s10 += wv * v10; s11 += wv * v11;
            }
        }
    }
    float bb = bias[c];
    float r00 = fmaxf(s00 + bb, 0.f), r01 = fmaxf(s01 + bb, 0.f);
    float r10 = fmaxf(s10 + bb, 0.f), r11 = fmaxf(s11 + bb, 0.f);
    float m = fmaxf(fmaxf(r00, r01), fmaxf(r10, r11));
    h2[(size_t)b * KPAD + c * 3721 + yh * 61 + xw] = (_Float16)m;
}

// Zero the K padding region of h2 (elements 55815..55839 of each row).
__global__ void k_pad_h2(_Float16* __restrict__ h2) {
    int idx = blockIdx.x * blockDim.x + threadIdx.x;  // 256*32
    int b = idx >> 5;
    int i = idx & 31;
    if (i < (KPAD - KFEAT)) h2[(size_t)b * KPAD + KFEAT + i] = (_Float16)0.f;
}

// fc1 weights f32[120,55815] -> f16[NPAD,KPAD] zero padded.
__global__ void k_cvt_w1(const float* __restrict__ w, _Float16* __restrict__ o) {
    size_t idx = (size_t)blockIdx.x * blockDim.x + threadIdx.x; // NPAD*KPAD
    int n = (int)(idx / KPAD);
    int k = (int)(idx % KPAD);
    float v = (n < 120 && k < KFEAT) ? w[(size_t)n * KFEAT + k] : 0.f;
    o[idx] = (_Float16)v;
}

// ---------------- fc1 GEMM via WMMA f32_16x16x32_f16 ----------------------
// C[256,128] = relu(A[256,KPAD] * B[128,KPAD]^T + bias).  One wave per tile.
// Async path: each 16x32 f16 tile (1 KB) is copied global->LDS with two
// global_load_async_to_lds_b128 ops (32 lanes x 16B each), double buffered,
// synchronized with s_wait_asynccnt; fragments then read via ds_load_b128.
// A-frag: lanes 0-15 take K{0..7,16..23}, lanes 16-31 take K{8..15,24..31}.
// B-frag: lane N=l takes K{0..7,16..23} (lanes<16) or K{8..15,24..31} (lanes>=16).
__global__ void k_fc1_wmma(const _Float16* __restrict__ A,
                           const _Float16* __restrict__ Bw,
                           const float* __restrict__ bias,
                           float* __restrict__ C) {
    int lane = threadIdx.x;           // 0..31
    int nt = blockIdx.x;              // 0..7   (N tiles)
    int mt = blockIdx.y;              // 0..15  (M tiles)
    int half = lane >> 4;
    int l = lane & 15;
    const int aoff0 = half ? 8 : 0;
    const int aoff1 = half ? 24 : 16;
    const int boff  = half ? 8 : 0;

    v8f acc = {};

#if HAS_ASYNC_LDS
    __shared__ __align__(16) _Float16 sA[2][512];   // 2 x (16 rows x 32 f16)
    __shared__ __align__(16) _Float16 sB[2][512];
    const _Float16* ag = A  + (size_t)(mt * 16) * KPAD;
    const _Float16* bg = Bw + (size_t)(nt * 16) * KPAD;
    // tile = 64 chunks of 16B; lane covers chunks {lane, lane+32}
    const int c0 = lane,      r0 = c0 >> 2, q0 = c0 & 3;
    const int c1 = lane + 32, r1 = c1 >> 2, q1 = c1 & 3;
    const int NSTEP = KPAD / 32;      // 1745

    // prologue: fill buffer 0 with tile k=0
    async_cp16(ag + (size_t)r0 * KPAD + q0 * 8, &sA[0][c0 * 8]);
    async_cp16(ag + (size_t)r1 * KPAD + q1 * 8, &sA[0][c1 * 8]);
    async_cp16(bg + (size_t)r0 * KPAD + q0 * 8, &sB[0][c0 * 8]);
    async_cp16(bg + (size_t)r1 * KPAD + q1 * 8, &sB[0][c1 * 8]);

    for (int kt = 0; kt < NSTEP; ++kt) {
        int cur = kt & 1;
        if (kt + 1 < NSTEP) {
            int nxt = cur ^ 1;
            int k = (kt + 1) * 32;
            async_cp16(ag + (size_t)r0 * KPAD + k + q0 * 8, &sA[nxt][c0 * 8]);
            async_cp16(ag + (size_t)r1 * KPAD + k + q1 * 8, &sA[nxt][c1 * 8]);
            async_cp16(bg + (size_t)r0 * KPAD + k + q0 * 8, &sB[nxt][c0 * 8]);
            async_cp16(bg + (size_t)r1 * KPAD + k + q1 * 8, &sB[nxt][c1 * 8]);
            // in-order completion: oldest 4 (current tile) are done
            asm volatile("s_wait_asynccnt 0x4" ::: "memory");
        } else {
            asm volatile("s_wait_asynccnt 0x0" ::: "memory");
        }
        FragF16 a, b;
        const _Float16* la = &sA[cur][0];
        const _Float16* lb = &sB[cur][0];
        a.q[0] = *(const u32x4*)(la + l * 32 + aoff0);   // ds_load_b128
        a.q[1] = *(const u32x4*)(la + l * 32 + aoff1);
        b.q[0] = *(const u32x4*)(lb + l * 32 + boff);
        b.q[1] = *(const u32x4*)(lb + l * 32 + boff + 16);
        acc = __builtin_amdgcn_wmma_f32_16x16x32_f16(
            false, a.h, false, b.h, (short)0, acc, false, false);
    }
#else
    // fallback: direct global b128 fragment loads
    const _Float16* arow = A  + (size_t)(mt * 16 + l) * KPAD;
    const _Float16* brow = Bw + (size_t)(nt * 16 + l) * KPAD;
    for (int k = 0; k < KPAD; k += 32) {
        if (k + 64 < KPAD) {
            __builtin_prefetch(arow + k + 64, 0, 3);
            __builtin_prefetch(brow + k + 64, 0, 3);
        }
        FragF16 a, b;
        a.q[0] = *(const u32x4*)(arow + k + aoff0);
        a.q[1] = *(const u32x4*)(arow + k + aoff1);
        b.q[0] = *(const u32x4*)(brow + k + boff);
        b.q[1] = *(const u32x4*)(brow + k + boff + 16);
        acc = __builtin_amdgcn_wmma_f32_16x16x32_f16(
            false, a.h, false, b.h, (short)0, acc, false, false);
    }
#endif

    // D layout: N = nt*16 + l, M = mt*16 + (half ? r+8 : r)
    int n = nt * 16 + l;
    if (n < 120) {
        float bb = bias[n];
        #pragma unroll
        for (int r = 0; r < 8; ++r) {
            int m = mt * 16 + (half ? r + 8 : r);
            float v = acc[r] + bb;
            C[(size_t)m * NPAD + n] = v > 0.f ? v : 0.f;
        }
    }
}

// ---------------- fc2 + norms + fc3 + parameterized head ------------------
// one block (96 threads) per batch element.
__global__ void k_fc2_head(const float* __restrict__ h3,
                           const float* __restrict__ fc2_w,
                           const float* __restrict__ fc2_b,
                           const float* __restrict__ fc3_w,
                           const float* __restrict__ fc3_b,
                           const float* __restrict__ pl_w,
                           const float* __restrict__ pl_b,
                           const float* __restrict__ pl_scale,
                           const float* __restrict__ pl_bias,
                           float* __restrict__ features,
                           float* __restrict__ norms,
                           float* __restrict__ probs) {
    int b = blockIdx.x;
    int n = threadIdx.x;
    __shared__ float f[84];
    if (n < 84) {
        const float* hp = h3 + (size_t)b * NPAD;
        const float* wp = fc2_w + n * 120;
        float s = fc2_b[n];
        for (int k = 0; k < 120; ++k) s += hp[k] * wp[k];
        f[n] = s;
        features[b * 84 + n] = s;
    }
    __syncthreads();
    if (n == 0) {
        float ss = 0.f, lg = fc3_b[0];
        for (int i = 0; i < 84; ++i) { ss += f[i] * f[i]; lg += f[i] * fc3_w[i]; }
        norms[b] = sqrtf(ss);
        float z = tanhf(lg * pl_w[0] + pl_b[0]) * pl_scale[0] + pl_bias[0];
        probs[b] = 1.f / (1.f + expf(-z));
    }
}

// ---------------- cosine graph + degree-weighted aggregation --------------
// one block (256 threads) per row b; thread j handles neighbor j.
__global__ void k_graph_agg(const float* __restrict__ features,
                            const float* __restrict__ norms,
                            const float* __restrict__ probs,
                            float* __restrict__ out) {
    int b = blockIdx.x;
    int j = threadIdx.x;
    __shared__ float fb[84];
    __shared__ float sdeg[256];
    __shared__ float swsum[256];
    if (j < 84) fb[j] = features[b * 84 + j];
    __syncthreads();

    const float* fj = features + j * 84;
    float dot = 0.f;
    for (int i = 0; i < 84; ++i) dot += fb[i] * fj[i];
    float sim = dot / (norms[b] * norms[j] + 1e-12f);
    float a = (sim >= 0.0f && j != b) ? 1.f : 0.f;   // THRESH = 0.0
    sdeg[j]  = a;
    swsum[j] = a * probs[j];
    __syncthreads();
    for (int st = 128; st > 0; st >>= 1) {
        if (j < st) { sdeg[j] += sdeg[j + st]; swsum[j] += swsum[j + st]; }
        __syncthreads();
    }
    if (j == 0) {
        float deg = sdeg[0], ws = swsum[0];
        float nm  = deg > 0.f ? ws / fmaxf(deg, 1.f) : 0.f;
        float agg = (probs[b] + nm) / (1.f + deg);
        out[b * 2 + 0] = agg;
        out[b * 2 + 1] = 1.f - agg;
    }
}

// ---------------------------------------------------------------------------
extern "C" void kernel_launch(void* const* d_in, const int* in_sizes, int n_in,
                              void* d_out, int out_size, void* d_ws, size_t ws_size,
                              hipStream_t stream) {
    const float* x     = (const float*)d_in[0];
    const float* c1w   = (const float*)d_in[1];
    const float* c1b   = (const float*)d_in[2];
    const float* c2w   = (const float*)d_in[3];
    const float* c2b   = (const float*)d_in[4];
    const float* fc1w  = (const float*)d_in[5];
    const float* fc1b  = (const float*)d_in[6];
    const float* fc2w  = (const float*)d_in[7];
    const float* fc2b  = (const float*)d_in[8];
    const float* fc3w  = (const float*)d_in[9];
    const float* fc3b  = (const float*)d_in[10];
    const float* plw   = (const float*)d_in[11];
    const float* plb   = (const float*)d_in[12];
    const float* pls   = (const float*)d_in[13];
    const float* plbi  = (const float*)d_in[14];

    char* ws = (char*)d_ws;
    const size_t o_h1   = 0;                         // 256*6*123*123 f16 = 46,476,288
    const size_t o_h2   = o_h1 + 46476288;           // 256*KPAD f16     = 28,590,080
    const size_t o_w1   = o_h2 + 28590080;           // NPAD*KPAD f16    = 14,295,040
    const size_t o_h3   = o_w1 + 14295040;           // 256*128 f32      = 131,072
    const size_t o_feat = o_h3 + 131072;             // 256*84 f32       = 86,016
    const size_t o_norm = o_feat + 86016;            // 1,024
    const size_t o_prob = o_norm + 1024;             // 1,024

    _Float16* h1    = (_Float16*)(ws + o_h1);
    _Float16* h2    = (_Float16*)(ws + o_h2);
    _Float16* w1    = (_Float16*)(ws + o_w1);
    float*    h3    = (float*)(ws + o_h3);
    float*    feat  = (float*)(ws + o_feat);
    float*    norms = (float*)(ws + o_norm);
    float*    probs = (float*)(ws + o_prob);

    k_conv1_pool<<<90774, 256, 0, stream>>>(x, c1w, c1b, h1);
    k_conv2_pool<<<55815, 256, 0, stream>>>(h1, c2w, c2b, h2);
    k_pad_h2<<<32, 256, 0, stream>>>(h2);
    k_cvt_w1<<<27920, 256, 0, stream>>>(fc1w, w1);
    k_fc1_wmma<<<dim3(8, 16), 32, 0, stream>>>(h2, w1, fc1b, h3);
    k_fc2_head<<<256, 96, 0, stream>>>(h3, fc2w, fc2b, fc3w, fc3b,
                                       plw, plb, pls, plbi, feat, norms, probs);
    k_graph_agg<<<256, 256, 0, stream>>>(feat, norms, probs, (float*)d_out);
}
